// Involution_bin_75282186764559
// MI455X (gfx1250) — matrix operand order repaired and checked by reading, not compiled, not run on hardware
//
#include <hip/hip_runtime.h>

typedef __attribute__((ext_vector_type(16))) _Float16 v16h;
typedef __attribute__((ext_vector_type(8)))  float    v8f;

#define EPSF 1e-5f
constexpr int BATCH = 8, CH = 64, HDIM = 128, WDIM = 128;
constexpr int HW = HDIM * WDIM;                     // 16384
constexpr size_t TEN = (size_t)BATCH * CH * HW;     // 8388608 elements (32 MiB)
constexpr float NRED = (float)(BATCH * HW);         // BN reduction count per channel

__device__ __forceinline__ float signf(float v) {
  return (v > 0.f) ? 1.f : ((v < 0.f) ? -1.f : 0.f);
}

// ---------------------------------------------------------------------------
// K0: fold binary weights (sf * sign), build M9 = span_f @ reduce_f, zero stats
// ---------------------------------------------------------------------------
__global__ void prep_kernel(const float* __restrict__ pre_w, const float* __restrict__ post_w,
                            const float* __restrict__ reduce_w, const float* __restrict__ span_w,
                            float* __restrict__ pre_sf, float* __restrict__ post_sf,
                            _Float16* __restrict__ pre_sgn, _Float16* __restrict__ post_sgn,
                            float* __restrict__ M9, float* __restrict__ stats_all) {
  __shared__ float spanf[9 * 32];
  __shared__ float sfred[32];
  const int t = threadIdx.x;  // 64 threads
  {
    float s = 0.f;
    for (int i = 0; i < 64; ++i) s += fabsf(pre_w[t * 64 + i]);
    pre_sf[t] = s * (1.f / 64.f);
    for (int i = 0; i < 64; ++i) pre_sgn[t * 64 + i] = (_Float16)signf(pre_w[t * 64 + i]);
    s = 0.f;
    for (int i = 0; i < 64; ++i) s += fabsf(post_w[t * 64 + i]);
    post_sf[t] = s * (1.f / 64.f);
    for (int i = 0; i < 64; ++i) post_sgn[t * 64 + i] = (_Float16)signf(post_w[t * 64 + i]);
  }
  if (t < 9) {
    float s = 0.f;
    for (int r = 0; r < 32; ++r) s += fabsf(span_w[t * 32 + r]);
    s *= (1.f / 32.f);
    for (int r = 0; r < 32; ++r) spanf[t * 32 + r] = s * signf(span_w[t * 32 + r]);
  }
  if (t < 32) {
    float s = 0.f;
    for (int c = 0; c < 64; ++c) s += fabsf(reduce_w[t * 64 + c]);
    sfred[t] = s * (1.f / 64.f);
  }
  __syncthreads();
  for (int k = 0; k < 9; ++k) {
    float acc = 0.f;
    for (int r = 0; r < 32; ++r)
      acc = fmaf(spanf[k * 32 + r], sfred[r] * signf(reduce_w[r * 64 + t]), acc);
    M9[k * 64 + t] = acc;
  }
  for (int i = t; i < 384; i += 64) stats_all[i] = 0.f;  // zero 6 x 64 stat slots
}

// ---------------------------------------------------------------------------
// WMMA GEMM: out[b, m, pix] = sf[m] * sum_k sgn[m,k] * transform(in[b, k, pix])
// MODE 0 (pre):  transform = sign(v) + bias[k]
// MODE 1 (post): transform = max(affA[k]*v + affB[k], 0) + bias[k]
// A operands staged through LDS in WMMA lane order, then held in registers for
// the whole tile loop. __launch_bounds__(256, 1) lifts the VGPR cap so the
// ~64-VGPR A live range does not spill to scratch (seen in prior rounds).
// ---------------------------------------------------------------------------
template <int MODE>
__global__ void __launch_bounds__(256, 1)
gemm_sign_kernel(const float* __restrict__ in,
                 const _Float16* __restrict__ sgn,
                 const float* __restrict__ sf,
                 const float* __restrict__ bias,
                 const float* __restrict__ affA,
                 const float* __restrict__ affB,
                 float* __restrict__ out) {
  // [operand o = t*2+kt][lane][16 halves], 32 B per (o, lane): 8 KB total
  __shared__ __attribute__((aligned(32))) _Float16 aswz[8 * 32 * 16];

  const int tid = threadIdx.x;
  const int lane = tid & 31;
  const int lrow = lane & 15;
  const int hb = lane >> 4;

  // Cooperative swizzled fill: 256 threads == 8 operands x 32 lanes, one each.
  {
    const int o = tid >> 5;          // operand id 0..7
    const int l = tid & 31;
    const int t = o >> 1, kt = o & 1;
    const int m = t * 16 + (l & 15);
    const int kb = kt * 32 + (l >> 4) * 8;
    _Float16* dst = &aswz[(o * 32 + l) * 16];
#pragma unroll
    for (int j = 0; j < 16; ++j) {
      const int k = kb + j + ((j >= 8) ? 8 : 0);  // ISA 16-bit A layout
      dst[j] = sgn[m * 64 + k];
    }
  }
  __syncthreads();

  // Explicit register preload of all 8 A operands (held across the loop).
  v16h Ar[4][2];
#pragma unroll
  for (int t = 0; t < 4; ++t)
#pragma unroll
    for (int kt = 0; kt < 2; ++kt)
      Ar[t][kt] = *(const v16h*)&aswz[((t * 2 + kt) * 32 + lane) * 16];

  const int wid = (blockIdx.x * blockDim.x + tid) >> 5;
  const int nwaves = (gridDim.x * blockDim.x) >> 5;

  for (int tile = wid; tile < BATCH * (HW / 16); tile += nwaves) {
    const int b = tile >> 10;
    const int n0 = (tile & 1023) << 4;
    const int nn = n0 + lrow;
    const float* inb = in + (size_t)b * CH * HW;

    // B operands (same k order as A: lanes 0-15 K=0..7,16..23, lanes 16-31 K=8..15,24..31)
    v16h Bv[2];
#pragma unroll
    for (int kt = 0; kt < 2; ++kt) {
      const int kb = kt * 32 + hb * 8;
#pragma unroll
      for (int j = 0; j < 16; ++j) {
        const int k = kb + j + ((j >= 8) ? 8 : 0);
        const float v = inb[(size_t)k * HW + nn];
        float g;
        if (MODE == 0) {
          g = signf(v) + bias[k];
        } else {
          g = fmaxf(fmaf(affA[k], v, affB[k]), 0.f) + bias[k];
        }
        Bv[kt][j] = (_Float16)g;
      }
    }

    float* outb = out + (size_t)b * CH * HW;
#pragma unroll
    for (int t = 0; t < 4; ++t) {
      v8f c = {};
      c = __builtin_amdgcn_wmma_f32_16x16x32_f16(false, Ar[t][0], false, Bv[0],
                                                 (short)0, c, false, false);
      c = __builtin_amdgcn_wmma_f32_16x16x32_f16(false, Ar[t][1], false, Bv[1],
                                                 (short)0, c, false, false);
#pragma unroll
      for (int r = 0; r < 8; ++r) {
        const int m = t * 16 + r + hb * 8;  // ISA C/D layout
        outb[(size_t)m * HW + nn] = c[r] * sf[m];
      }
    }
  }
}

// ---------------------------------------------------------------------------
// Per-channel sum / sum-of-squares over one (b, c) slice per block
// ---------------------------------------------------------------------------
__global__ void stats_kernel(const float* __restrict__ t,
                             float* __restrict__ sum, float* __restrict__ sq) {
  __shared__ float rs[256], rq[256];
  const int tid = threadIdx.x;
  const int c = blockIdx.x & 63;
  const size_t base = (size_t)blockIdx.x * HW;
  float s = 0.f, q = 0.f;
  for (int i = tid; i < HW; i += 256) {
    const float v = t[base + i];
    s += v;
    q = fmaf(v, v, q);
  }
  rs[tid] = s; rq[tid] = q;
  __syncthreads();
  for (int off = 128; off > 0; off >>= 1) {
    if (tid < off) { rs[tid] += rs[tid + off]; rq[tid] += rq[tid + off]; }
    __syncthreads();
  }
  if (tid == 0) { atomicAdd(&sum[c], rs[0]); atomicAdd(&sq[c], rq[0]); }
}

// Fold BN stats into per-channel affine: a = gamma*rsqrt(var+eps), b = beta - a*mean
__global__ void bnaff_kernel(const float* __restrict__ sum, const float* __restrict__ sq,
                             const float* __restrict__ gamma, const float* __restrict__ beta,
                             float* __restrict__ a, float* __restrict__ bc) {
  const int t = threadIdx.x;
  if (t < CH) {
    const float m = sum[t] / NRED;
    const float v = sq[t] / NRED - m * m;
    const float ai = gamma[t] * rsqrtf(v + EPSF);
    a[t] = ai;
    bc[t] = beta[t] - ai * m;
  }
}

// ---------------------------------------------------------------------------
// Fused: h = prelu(BN(y)) + mid_bias (recomputed from y via folded affine),
// ker = M9 @ h (registers), involution z = sum_k h[3x3 nbr] * ker[k]
// 16x16 spatial tile per block, batch from block index, loop over all channels, LDS halo tile
// ---------------------------------------------------------------------------
__global__ void involution_kernel(const float* __restrict__ y,
                                  const float* __restrict__ a0, const float* __restrict__ b0,
                                  const float* __restrict__ pre_a,
                                  const float* __restrict__ mid_bias,
                                  const float* __restrict__ M9,
                                  float* __restrict__ z) {
  __shared__ float hs[18 * 18];
  __shared__ float m9s[9 * 64];
  const int tid = threadIdx.x;
  const int blk = blockIdx.x;           // 512 blocks = 8 batches x 8x8 tiles
  const int b = blk >> 6;
  const int ty = (blk >> 3) & 7, tx = blk & 7;
  const int py = ty * 16 + (tid >> 4);
  const int px = tx * 16 + (tid & 15);
  for (int i = tid; i < 9 * 64; i += 256) m9s[i] = M9[i];
  __syncthreads();

  const float* yb = y + (size_t)b * CH * HW;
  float ker[9];
#pragma unroll
  for (int k = 0; k < 9; ++k) ker[k] = 0.f;
  for (int c = 0; c < CH; ++c) {
    const float v = yb[(size_t)c * HW + py * WDIM + px];
    const float t = fmaf(a0[c], v, b0[c]);
    const float h = fmaxf(t, 0.f) + pre_a[c] * fminf(t, 0.f) + mid_bias[c];
#pragma unroll
    for (int k = 0; k < 9; ++k) ker[k] = fmaf(m9s[k * 64 + c], h, ker[k]);
  }

  float* zb = z + (size_t)b * CH * HW;
  const int ly = (tid >> 4) + 1, lx = (tid & 15) + 1;
  for (int c = 0; c < CH; ++c) {
    __syncthreads();  // protect hs from previous iteration's readers
    const float ac = a0[c], bcf = b0[c], pc = pre_a[c], mb = mid_bias[c];
    for (int e = tid; e < 18 * 18; e += 256) {
      const int ey = e / 18, ex = e % 18;
      const int gy = ty * 16 + ey - 1, gx = tx * 16 + ex - 1;
      float h = 0.f;  // zero padding
      if (gy >= 0 && gy < HDIM && gx >= 0 && gx < WDIM) {
        const float v = yb[(size_t)c * HW + gy * WDIM + gx];
        const float t = fmaf(ac, v, bcf);
        h = fmaxf(t, 0.f) + pc * fminf(t, 0.f) + mb;
      }
      hs[e] = h;
    }
    __syncthreads();
    float acc = 0.f;
#pragma unroll
    for (int i = 0; i < 3; ++i)
#pragma unroll
      for (int j = 0; j < 3; ++j)
        acc = fmaf(hs[(ly + i - 1) * 18 + (lx + j - 1)], ker[i * 3 + j], acc);
    zb[(size_t)c * HW + py * WDIM + px] = acc;
  }
}

// out = BN_post(p) + x   (ReLU-then-PReLU of mid stage is identity, folded out)
__global__ void final_kernel(const float* __restrict__ p,
                             const float* __restrict__ a2, const float* __restrict__ b2,
                             const float* __restrict__ x, float* __restrict__ out) {
  const size_t n4 = TEN / 4;
  for (size_t i = (size_t)blockIdx.x * blockDim.x + threadIdx.x; i < n4;
       i += (size_t)gridDim.x * blockDim.x) {
    const int c = (int)(((i * 4) >> 14) & 63);
    const float4 pv = reinterpret_cast<const float4*>(p)[i];
    const float4 xv = reinterpret_cast<const float4*>(x)[i];
    const float a = a2[c], b = b2[c];
    float4 o;
    o.x = fmaf(a, pv.x, b) + xv.x;
    o.y = fmaf(a, pv.y, b) + xv.y;
    o.z = fmaf(a, pv.z, b) + xv.z;
    o.w = fmaf(a, pv.w, b) + xv.w;
    reinterpret_cast<float4*>(out)[i] = o;
  }
}

extern "C" void kernel_launch(void* const* d_in, const int* in_sizes, int n_in,
                              void* d_out, int out_size, void* d_ws, size_t ws_size,
                              hipStream_t stream) {
  const float* x          = (const float*)d_in[0];
  const float* pre_bias   = (const float*)d_in[1];
  const float* pre_w      = (const float*)d_in[2];
  const float* pre_gamma  = (const float*)d_in[3];
  const float* pre_beta   = (const float*)d_in[4];
  const float* pre_a      = (const float*)d_in[5];
  const float* mid_bias   = (const float*)d_in[6];
  const float* reduce_w   = (const float*)d_in[7];
  const float* span_w     = (const float*)d_in[8];
  const float* mid_gamma  = (const float*)d_in[9];
  const float* mid_beta   = (const float*)d_in[10];
  // d_in[11] = mid_a: PReLU applied after ReLU -> provably identity, unused.
  const float* post_bias  = (const float*)d_in[12];
  const float* post_w     = (const float*)d_in[13];
  const float* post_gamma = (const float*)d_in[14];
  const float* post_beta  = (const float*)d_in[15];
  float* out = (float*)d_out;

  float* ws   = (float*)d_ws;
  float* buf0 = ws;              // y (pre-conv out), later reused as p (post-conv out)
  float* buf1 = ws + TEN;        // z (involution out)
  float* sm   = ws + 2 * TEN;    // small region
  float* pre_sf = sm, *post_sf = sm + 64, *M9 = sm + 128;
  float* st = sm + 704;          // 6 x 64 stats
  float* s0s = st, *s0q = st + 64, *s1s = st + 128, *s1q = st + 192,
       * s2s = st + 256, *s2q = st + 320;
  float* af = sm + 1088;         // 6 x 64 folded affines
  float* a0 = af, *b0 = af + 64, *a1 = af + 128, *b1 = af + 192,
       * a2 = af + 256, *b2 = af + 320;
  _Float16* pre_sgn  = (_Float16*)(sm + 1472);  // 64x64 f16
  _Float16* post_sgn = pre_sgn + 4096;          // 64x64 f16

  prep_kernel<<<1, 64, 0, stream>>>(pre_w, post_w, reduce_w, span_w,
                                    pre_sf, post_sf, pre_sgn, post_sgn, M9, st);
  gemm_sign_kernel<0><<<256, 256, 0, stream>>>(x, pre_sgn, pre_sf, pre_bias,
                                               nullptr, nullptr, buf0);
  stats_kernel<<<512, 256, 0, stream>>>(buf0, s0s, s0q);
  bnaff_kernel<<<1, 64, 0, stream>>>(s0s, s0q, pre_gamma, pre_beta, a0, b0);
  involution_kernel<<<512, 256, 0, stream>>>(buf0, a0, b0, pre_a, mid_bias, M9, buf1);
  stats_kernel<<<512, 256, 0, stream>>>(buf1, s1s, s1q);
  bnaff_kernel<<<1, 64, 0, stream>>>(s1s, s1q, mid_gamma, mid_beta, a1, b1);
  gemm_sign_kernel<1><<<256, 256, 0, stream>>>(buf1, post_sgn, post_sf, post_bias,
                                               a1, b1, buf0);
  stats_kernel<<<512, 256, 0, stream>>>(buf0, s2s, s2q);
  bnaff_kernel<<<1, 64, 0, stream>>>(s2s, s2q, post_gamma, post_beta, a2, b2);
  final_kernel<<<2048, 256, 0, stream>>>(buf0, a2, b2, x, out);
}
